// SEALModel_35141422416033
// MI455X (gfx1250) — matrix operand order repaired and checked by reading, not compiled, not run on hardware
//
#include <hip/hip_runtime.h>

#define G_NUM 8192
#define NPG 32
#define E_NUM 2097152
#define LABEL_DIM 32
#define HID 64
#define MAX_LABEL 50
#define WPB 8            // waves (graphs) per block
#define TPB (WPB * 32)   // wave32

typedef __attribute__((ext_vector_type(16))) _Float16 v16h;
typedef __attribute__((ext_vector_type(8)))  _Float16 v8h;
typedef __attribute__((ext_vector_type(8)))  float    v8f;

#define WMMA(a, b, c) __builtin_amdgcn_wmma_f32_16x16x32_f16(false, (a), false, (b), (short)0, (c), false, false)

static __device__ __forceinline__ v8f v8f_zero() {
  v8f z = {0.f, 0.f, 0.f, 0.f, 0.f, 0.f, 0.f, 0.f};
  return z;
}

// ---------------------------------------------------------------------------------------
// A 16x32 f16 fragment from row-major f16 LDS (ld parameterized):
// lane row = m0+(l&15); halves 0..15 = K kb..kb+15, kb = k0+16*(l>=16)
static __device__ __forceinline__ v16h fetch_A_ld(const _Float16* X, int ld, int m0, int k0, int lane) {
  const _Float16* row = X + (m0 + (lane & 15)) * ld + k0 + ((lane >> 4) << 4);
  v8h lo = *(const v8h*)row;
  v8h hi = *(const v8h*)(row + 8);
  v16h a;
#pragma unroll
  for (int e = 0; e < 8; ++e) { a[e] = lo[e]; a[e + 8] = hi[e]; }
  return a;
}

// Pre-swizzled fragment (A- or B-layout): 32 lanes x 16 contiguous halves.
static __device__ __forceinline__ v16h fetch_frag(const _Float16* sW, int frag, int lane) {
  return *(const v16h*)(sW + frag * 512 + lane * 16);
}

// B 32x16 fragment read contiguously from row-major [n][k] storage (ld halves per n-row).
static __device__ __forceinline__ v16h fetch_B_rm(const _Float16* Z, int ld, int n0, int k0, int lane) {
  return *(const v16h*)(Z + (n0 + (lane & 15)) * ld + k0 + ((lane >> 4) << 4));
}

// B fragment of (I+M)^T straight from global adjacency: B[k][n] = (I+M)[n][k] = adj row n.
// lane col n = nt*16+(l&15); halves e -> k = 16*(l>=16)+e : 16 consecutive ints.
static __device__ __forceinline__ v16h load_MT_frag(const int* ag, int nt, int lane) {
  int n = nt * 16 + (lane & 15);
  int kb = (lane >> 4) << 4;
  const int4* rp4 = (const int4*)(ag + n * NPG + kb);
  int4 c0 = rp4[0], c1 = rp4[1], c2 = rp4[2], c3 = rp4[3];
  int v[16] = {c0.x, c0.y, c0.z, c0.w, c1.x, c1.y, c1.z, c1.w,
               c2.x, c2.y, c2.z, c2.w, c3.x, c3.y, c3.z, c3.w};
  v16h b;
#pragma unroll
  for (int e = 0; e < 16; ++e)
    b[e] = (_Float16)(float)(v[e] + ((kb + e) == n ? 1 : 0));
  return b;
}

// ---------------------------------------------------------------------------------------
// Cooperative weight swizzlers (once per block).
// B-fragment order for W[KxHID]: frag f=kt*4+nt; elem(lane,e) = W[kt*32+16*(l>=16)+e][nt*16+(l&15)]
template <int K>
static __device__ void load_WB_frags(const float* __restrict__ W, _Float16* sW, int tid) {
  const int total = K * HID;
  for (int i = tid; i < total; i += TPB) {
    int f = i >> 9, r = i & 511;
    int lane = r >> 4, e = r & 15;
    int kt = f >> 2, nt = f & 3;
    int k = kt * 32 + ((lane >> 4) << 4) + e;
    int n = nt * 16 + (lane & 15);
    sW[i] = (_Float16)W[k * HID + n];
  }
}

// A-fragment order for W^T (64 x K): frag f = mt*(K/32)+kt;
// elem(lane,e): row = mt*16+(l&15); k = kt*32 + 16*(l>=16) + e; val = W[k][row]
template <int K>
static __device__ void load_WT_frags(const float* __restrict__ W, _Float16* sW, int tid) {
  const int total = K * HID;
  for (int i = tid; i < total; i += TPB) {
    int f = i >> 9, r = i & 511;
    int lane = r >> 4, e = r & 15;
    int mt = f / (K / 32), kt = f % (K / 32);
    int row = mt * 16 + (lane & 15);
    int k = kt * 32 + ((lane >> 4) << 4) + e;
    sW[i] = (_Float16)W[k * HID + row];
  }
}

// ---------------------------------------------------------------------------------------
// Swapped aggregation: Zr = ((I+M) h) stored row-major [node][feat], computed as h^T (I+M)^T.
// A = h^T from T storage [feat][node] (ld 32, contiguous); B = (I+M)^T register frags.
// D: lane holds node fixed, 8 consecutive features -> one b128 store into Zr.
template <int FT>   // feature tiles: F/16 (2 or 4)
static __device__ __forceinline__ void stage_agg_T(const _Float16* hT, v16h bM0, v16h bM1,
                                                   _Float16* Zr, int lane) {
#pragma unroll
  for (int mt = 0; mt < FT; ++mt) {
    v8f acc0 = v8f_zero(), acc1 = v8f_zero();
    v16h a = fetch_A_ld(hT, NPG, mt * 16, 0, lane);
    acc0 = WMMA(a, bM0, acc0);
    acc1 = WMMA(a, bM1, acc1);
    int rbase = mt * 16 + ((lane >> 4) << 3);   // feature base
    int n_lo = lane & 15;                       // node within tile
#pragma unroll
    for (int nt = 0; nt < 2; ++nt) {
      v8f acc = nt ? acc1 : acc0;
      v8h o;
#pragma unroll
      for (int r = 0; r < 8; ++r) o[r] = (_Float16)acc[r];
      *(v8h*)(Zr + (nt * 16 + n_lo) * HID + rbase) = o;
    }
  }
}

// Swapped MLP: Yr = relu(Zr @ W + b) computed as W^T Z^T; A = W^T frags, B = Zr contiguous.
// Output row-major, contiguous b128 stores. In-place (Yr==Zr) safe: all B reads precede writes.
template <int K>
static __device__ __forceinline__ void stage_mlp_T(const _Float16* Zr, const _Float16* sWT,
                                                   const float* bias, _Float16* Yr, int lane) {
  v8f acc[4][2];
#pragma unroll
  for (int mt = 0; mt < 4; ++mt) { acc[mt][0] = v8f_zero(); acc[mt][1] = v8f_zero(); }
#pragma unroll
  for (int kt = 0; kt < K / 32; ++kt) {
    v16h b[2];
#pragma unroll
    for (int nt = 0; nt < 2; ++nt) b[nt] = fetch_B_rm(Zr, HID, nt * 16, kt * 32, lane);
#pragma unroll
    for (int mt = 0; mt < 4; ++mt) {
      v16h a = fetch_frag(sWT, mt * (K / 32) + kt, lane);
#pragma unroll
      for (int nt = 0; nt < 2; ++nt) acc[mt][nt] = WMMA(a, b[nt], acc[mt][nt]);
    }
  }
#pragma unroll
  for (int mt = 0; mt < 4; ++mt) {
    int rbase = mt * 16 + ((lane >> 4) << 3);   // out-feature base
    int n_lo = lane & 15;
#pragma unroll
    for (int nt = 0; nt < 2; ++nt) {
      v8h o;
#pragma unroll
      for (int r = 0; r < 8; ++r)
        o[r] = (_Float16)fmaxf(acc[mt][nt][r] + bias[rbase + r], 0.f);
      *(v8h*)(Yr + (nt * 16 + n_lo) * HID + rbase) = o;
    }
  }
}

// Standard MLP with transposed output: YT[feat][node] = relu(Xr @ W + b).
// A from row-major (contiguous); B = weight frags; D's 8 node-rows per lane are contiguous in YT.
static __device__ __forceinline__ void stage_mlp_std_T(const _Float16* Xr, const _Float16* sW,
                                                       const float* bias, _Float16* YT, int lane) {
  v8f acc[2][4];
#pragma unroll
  for (int mt = 0; mt < 2; ++mt)
#pragma unroll
    for (int nt = 0; nt < 4; ++nt) acc[mt][nt] = v8f_zero();
#pragma unroll
  for (int kt = 0; kt < 2; ++kt) {
    v16h b[4];
#pragma unroll
    for (int nt = 0; nt < 4; ++nt) b[nt] = fetch_frag(sW, kt * 4 + nt, lane);
#pragma unroll
    for (int mt = 0; mt < 2; ++mt) {
      v16h a = fetch_A_ld(Xr, HID, mt * 16, kt * 32, lane);
#pragma unroll
      for (int nt = 0; nt < 4; ++nt) acc[mt][nt] = WMMA(a, b[nt], acc[mt][nt]);
    }
  }
#pragma unroll
  for (int mt = 0; mt < 2; ++mt) {
    int rbase = mt * 16 + ((lane >> 4) << 3);   // node base
#pragma unroll
    for (int nt = 0; nt < 4; ++nt) {
      int c = nt * 16 + (lane & 15);            // out-feature
      v8h o;
#pragma unroll
      for (int r = 0; r < 8; ++r)
        o[r] = (_Float16)fmaxf(acc[mt][nt][r] + bias[c], 0.f);
      *(v8h*)(YT + c * NPG + rbase) = o;
    }
  }
}

// ---------------------------------------------------------------------------------------
__global__ void seal_zero_adj(int* __restrict__ adj, int n) {
  int i = blockIdx.x * blockDim.x + threadIdx.x;
  if (i < n) adj[i] = 0;
}

__global__ void seal_build_adj(const int* __restrict__ src, const int* __restrict__ dst,
                               int* __restrict__ adj) {
  int e = blockIdx.x * blockDim.x + threadIdx.x;
  if (e < E_NUM) {
    int s = src[e], d = dst[e];
    atomicAdd(&adj[((d >> 5) << 10) + ((d & 31) << 5) + (s & 31)], 1);
  }
}

__global__ void __launch_bounds__(TPB, 1)
seal_forward(const int* __restrict__ labels, const int* __restrict__ adj,
             const float* __restrict__ emb,
             const float* __restrict__ w0a, const float* __restrict__ b0a,
             const float* __restrict__ w0b, const float* __restrict__ b0b,
             const float* __restrict__ w1a, const float* __restrict__ b1a,
             const float* __restrict__ w1b, const float* __restrict__ b1b,
             const float* __restrict__ ws1, const float* __restrict__ bs1,
             const float* __restrict__ ws2, const float* __restrict__ bs2,
             float* __restrict__ out) {
  __shared__ __align__(32) _Float16 sW0aT[LABEL_DIM * HID];  // A-frags (W^T), 4 frags
  __shared__ __align__(32) _Float16 sW0b [HID * HID];        // B-frags,        8 frags
  __shared__ __align__(32) _Float16 sW1aT[HID * HID];        // A-frags (W^T),  8 frags
  __shared__ __align__(32) _Float16 sW1bT[HID * HID];        // A-frags (W^T),  8 frags
  __shared__ float sWs1[HID * HID];
  __shared__ float sB0a[HID], sB0b[HID], sB1a[HID], sB1b[HID], sBs1v[HID], sWs2v[HID];
  __shared__ __align__(32) _Float16 bufR[WPB][NPG * HID];    // row-major [node][feat]
  __shared__ __align__(32) _Float16 bufT[WPB][HID * NPG];    // transposed [feat][node]
  __shared__ float sHg[WPB][HID];

  const int tid = threadIdx.x;

  load_WT_frags<LABEL_DIM>(w0a, sW0aT, tid);
  load_WB_frags<HID>(w0b, sW0b, tid);
  load_WT_frags<HID>(w1a, sW1aT, tid);
  load_WT_frags<HID>(w1b, sW1bT, tid);
  for (int i = tid; i < HID * HID; i += TPB) sWs1[i] = ws1[i];
  if (tid < HID) {
    sB0a[tid] = b0a[tid];  sB0b[tid] = b0b[tid];
    sB1a[tid] = b1a[tid];  sB1b[tid] = b1b[tid];
    sBs1v[tid] = bs1[tid]; sWs2v[tid] = ws2[tid];
  }

  const int w = tid >> 5;
  const int lane = tid & 31;
  const int g = blockIdx.x * WPB + w;
  _Float16* R = bufR[w];
  _Float16* T = bufT[w];

  // (I + M)^T B-fragments straight to registers (16 consecutive ints per lane)
  const int* ag = adj + g * (NPG * NPG);
  v16h bM0 = load_MT_frag(ag, 0, lane);
  v16h bM1 = load_MT_frag(ag, 1, lane);

  // Embedding lookup (padding_idx=0) -> transposed h0T[feat][node]
  int lab = labels[g * NPG + lane];
  lab = lab < 0 ? 0 : (lab > MAX_LABEL ? MAX_LABEL : lab);
  const float* er = emb + lab * LABEL_DIM;
#pragma unroll 8
  for (int c = 0; c < LABEL_DIM; ++c)
    T[c * NPG + lane] = (_Float16)((lab != 0) ? er[c] : 0.f);
  __syncthreads();

  // GIN layer 0
  stage_agg_T<2>(T, bM0, bM1, R, lane);            __syncthreads(); // z0 = (I+M)h0   (R)
  stage_mlp_T<32>(R, sW0aT, sB0a, R, lane);        __syncthreads(); // relu(z0 W0a+b) (R, in place)
  stage_mlp_std_T(R, sW0b, sB0b, T, lane);         __syncthreads(); // h1             (T)
  // GIN layer 1
  stage_agg_T<4>(T, bM0, bM1, R, lane);            __syncthreads(); // z1 = (I+M)h1   (R)
  stage_mlp_T<64>(R, sW1aT, sB1a, R, lane);        __syncthreads();
  stage_mlp_T<64>(R, sW1bT, sB1b, R, lane);        __syncthreads(); // h2             (R)

  // Mean pool over the 32 nodes
#pragma unroll
  for (int half = 0; half < 2; ++half) {
    int c = lane + half * 32;
    float s = 0.f;
#pragma unroll
    for (int r = 0; r < NPG; ++r) s += (float)R[r * HID + c];
    sHg[w][c] = s * (1.0f / NPG);
  }
  __syncthreads();

  // Scorer: relu(hg @ ws1 + bs1) @ ws2 + bs2
  float p = 0.f;
#pragma unroll
  for (int half = 0; half < 2; ++half) {
    int c = lane + half * 32;
    float t = sBs1v[c];
#pragma unroll 8
    for (int k = 0; k < HID; ++k) t = fmaf(sHg[w][k], sWs1[k * HID + c], t);
    t = fmaxf(t, 0.f);
    p = fmaf(t, sWs2v[c], p);
  }
#pragma unroll
  for (int off = 16; off > 0; off >>= 1) p += __shfl_xor(p, off, 32);
  if (lane == 0) out[g] = p + bs2[0];
}

// ---------------------------------------------------------------------------------------
extern "C" void kernel_launch(void* const* d_in, const int* in_sizes, int n_in,
                              void* d_out, int out_size, void* d_ws, size_t ws_size,
                              hipStream_t stream) {
  const int*   labels = (const int*)d_in[0];
  const int*   src    = (const int*)d_in[1];
  const int*   dst    = (const int*)d_in[2];
  // d_in[3] graph_ids implicit (node/32)
  const float* emb = (const float*)d_in[4];
  const float* w0a = (const float*)d_in[5];
  const float* b0a = (const float*)d_in[6];
  const float* w0b = (const float*)d_in[7];
  const float* b0b = (const float*)d_in[8];
  const float* w1a = (const float*)d_in[9];
  const float* b1a = (const float*)d_in[10];
  const float* w1b = (const float*)d_in[11];
  const float* b1b = (const float*)d_in[12];
  const float* ws1 = (const float*)d_in[13];
  const float* bs1 = (const float*)d_in[14];
  const float* ws2 = (const float*)d_in[15];
  const float* bs2 = (const float*)d_in[16];

  int* adj = (int*)d_ws;                       // 8192*1024 ints = 33.5 MB, L2-resident
  const int nadj = G_NUM * NPG * NPG;

  seal_zero_adj<<<(nadj + 255) / 256, 256, 0, stream>>>(adj, nadj);
  seal_build_adj<<<(E_NUM + 255) / 256, 256, 0, stream>>>(src, dst, adj);
  seal_forward<<<G_NUM / WPB, TPB, 0, stream>>>(labels, adj, emb,
                                                w0a, b0a, w0b, b0b,
                                                w1a, b1a, w1b, b1b,
                                                ws1, bs1, ws2, bs2,
                                                (float*)d_out);
}
